// PointConv_21620865368538
// MI455X (gfx1250) — matrix operand order repaired and checked by reading, not compiled, not run on hardware
//
#include <hip/hip_runtime.h>
#include <hip/hip_bf16.h>

// ---------------------------------------------------------------------------
// PointConv forward for MI455X (gfx1250, wave32, WMMA).
// Pipeline (all on `stream`, graph-capture safe, no host sync):
//   0  zero_stats      : clear atomic-reduction block (re-run safe)
//   1  kde             : s[b,n] = mean_j exp(-|pi-pj|^2/2), + global stats of s
//   2  smlp1 / 3 smlp2 : density MLP 1->8->8->1 with train-mode BN (analytic
//                        stats for the linear first layer, atomics for layer 2)
//   4  fps             : farthest point sampling, LDS-resident dist + argmax
//   5  knn             : top-32 neighbors per query (LDS point tiles)
//   6  gather          : rel/xg/sg -> h0 [pt][64] f16, rel f32, wnet-L1 stats
//   7  pack            : f16 weight packs (m_w1, m_w2, l_w -> [o][w*128+c])
//   8  mlp_gemm<64>    : mp1 = m_w1 * h0      (WMMA f16->f32, LDS weights)
//   9  bnrelu          : mh1 = relu(bn(mp1))           [pt][64]
//  10  mlp_gemm<128>   : mp2 = m_w2 * mh1     (WMMA)
//  11  bnrelu(T)       : mh2 = relu(bn(mp2)) transposed to [128][pt]
//  12  wnet2 / 13 wnet3: weight net 3->8->8->16 (tiny; VALU), wfin [16][pt]
//  14  agg_wmma        : per (b,m): 16x128 = (16xK)*(Kx128), K=32 == one WMMA
//  15  final_gemm_wmma : Y[4096x256] = agg[4096x2048] * Wt[2048x256] (WMMA)
//  16  output          : out = concat(q, relu(bn(Y)))   B x 259 x M  (f32)
// Precision: f16 operands / f32 accumulation on all matrix ops (inputs are
// O(0.1) magnitude), f32 everywhere else -> the MI455X reward path without
// abandoning the fp32 reference numerically.
// ---------------------------------------------------------------------------

#define BDIM 4
#define CDIM 64
#define NDIM 4096
#define MDIM 1024
#define KNB  32
#define NP   (BDIM * NDIM)                     // 16384 points
#define PTN  ((size_t)BDIM * MDIM * KNB)       // 131072 gathered points
#define BMN  (BDIM * MDIM)                     // 4096 (b,m) pairs
#define EPSF 1e-5f

typedef __attribute__((ext_vector_type(16))) _Float16 v16h;
typedef __attribute__((ext_vector_type(8)))  _Float16 half8;
typedef __attribute__((ext_vector_type(8)))  float    v8f;

__device__ __forceinline__ v16h load16h(const _Float16* lo, const _Float16* hi) {
  half8 a = *(const half8*)lo;        // 16B contiguous
  half8 b = *(const half8*)hi;        // 16B contiguous
  v16h r;
#pragma unroll
  for (int j = 0; j < 8; ++j) { r[j] = a[j]; r[8 + j] = b[j]; }
  return r;
}

// ---------------------------------------------------------------- stats block
// float slots inside ws[0..4096B): zeroed each launch.
#define ST_S    0      // 2:   sum, sumsq of s
#define ST_P2   2      // 16:  density-MLP layer2 pre-act (8 ch)
#define ST_WP1  18     // 16:  weight-net layer1 pre-act (8 ch)
#define ST_WP2  34     // 16:  weight-net layer2 pre-act (8 ch)
#define ST_MP1  50     // 128: m-net layer1 pre-act (64 ch)
#define ST_MP2  178    // 256: m-net layer2 pre-act (128 ch)
#define ST_Y    434    // 512: final layer pre-act (256 ch)

__global__ void zero_stats_kernel(float* st) {
  for (int i = threadIdx.x; i < 1024; i += 256) st[i] = 0.f;
}

// ---------------------------------------------------------------- 1: KDE
__global__ void kde_kernel(const float* __restrict__ x, float* __restrict__ s_raw,
                           float* __restrict__ st) {
  const int b = blockIdx.y;
  const int i = blockIdx.x * 256 + threadIdx.x;
  const float* px = x + (size_t)b * CDIM * NDIM;
  const float pix = px[i], piy = px[NDIM + i], piz = px[2 * NDIM + i];
  __shared__ float tx[256], ty[256], tz[256];
  __shared__ float bs[2];
  if (threadIdx.x < 2) bs[threadIdx.x] = 0.f;
  float acc = 0.f;
  for (int j0 = 0; j0 < NDIM; j0 += 256) {
    const int j = j0 + threadIdx.x;
    tx[threadIdx.x] = px[j];
    ty[threadIdx.x] = px[NDIM + j];
    tz[threadIdx.x] = px[2 * NDIM + j];
    __syncthreads();
#pragma unroll 8
    for (int t = 0; t < 256; ++t) {
      const float dx = pix - tx[t], dy = piy - ty[t], dz = piz - tz[t];
      acc += __expf(-0.5f * (dx * dx + dy * dy + dz * dz));   // bw = 1
    }
    __syncthreads();
  }
  const float s = acc * (1.f / NDIM);
  s_raw[b * NDIM + i] = s;
  atomicAdd(&bs[0], s);
  atomicAdd(&bs[1], s * s);
  __syncthreads();
  if (threadIdx.x == 0) { atomicAdd(&st[ST_S + 0], bs[0]); atomicAdd(&st[ST_S + 1], bs[1]); }
}

// ---------------------------------------------------------------- 2: density MLP L1+L2
__global__ void smlp1_kernel(const float* __restrict__ s_raw, const float* __restrict__ st,
                             const float* sw1, const float* sb1, const float* sg1, const float* sbb1,
                             const float* sw2, const float* sb2,
                             float* __restrict__ p2buf, float* __restrict__ p2stat) {
  const int id = blockIdx.x * 256 + threadIdx.x;   // NP
  const float mu = st[ST_S] * (1.f / NP);
  const float var = st[ST_S + 1] * (1.f / NP) - mu * mu;
  const float s = s_raw[id];
  float h1[8];
#pragma unroll
  for (int c = 0; c < 8; ++c) {
    const float w = sw1[c];
    // conv1 is linear in s -> BN stats of layer-1 pre-act are analytic.
    const float hn = (s - mu) * w * rsqrtf(w * w * var + EPSF) * sg1[c] + sbb1[c];
    h1[c] = fmaxf(hn, 0.f);
  }
  __shared__ float ls[16];
  if (threadIdx.x < 16) ls[threadIdx.x] = 0.f;
  __syncthreads();
#pragma unroll
  for (int o = 0; o < 8; ++o) {
    float p = sb2[o];
#pragma unroll
    for (int c = 0; c < 8; ++c) p += sw2[o * 8 + c] * h1[c];
    p2buf[(size_t)id * 8 + o] = p;
    atomicAdd(&ls[o], p);
    atomicAdd(&ls[8 + o], p * p);
  }
  __syncthreads();
  if (threadIdx.x < 16) atomicAdd(&p2stat[threadIdx.x], ls[threadIdx.x]);
}

__global__ void smlp2_kernel(const float* __restrict__ p2buf, const float* __restrict__ p2stat,
                             const float* sg2, const float* sbb2,
                             const float* sw3, const float* sb3,
                             float* __restrict__ sfin) {
  const int id = blockIdx.x * 256 + threadIdx.x;   // NP
  float acc = sb3[0];
#pragma unroll
  for (int c = 0; c < 8; ++c) {
    const float mu = p2stat[c] * (1.f / NP);
    const float var = p2stat[8 + c] * (1.f / NP) - mu * mu;
    float v = p2buf[(size_t)id * 8 + c];
    v = (v - mu) * rsqrtf(var + EPSF) * sg2[c] + sbb2[c];
    acc += sw3[c] * fmaxf(v, 0.f);
  }
  sfin[id] = 1.f / (1.f + __expf(-acc));
}

// ---------------------------------------------------------------- 4: FPS
__global__ void fps_kernel(const float* __restrict__ x, int* __restrict__ qidx,
                           float* __restrict__ qbuf) {
  const int b = blockIdx.x, t = threadIdx.x;         // 256 threads / batch
  const float* px = x + (size_t)b * CDIM * NDIM;
  __shared__ float dist[NDIM];                        // 16 KB LDS-resident
  __shared__ float rv[256];
  __shared__ int   ri[256];
  const float p0x = px[0], p0y = px[NDIM], p0z = px[2 * NDIM];
  for (int i = t; i < NDIM; i += 256) {
    const float dx = px[i] - p0x, dy = px[NDIM + i] - p0y, dz = px[2 * NDIM + i] - p0z;
    dist[i] = dx * dx + dy * dy + dz * dz;
  }
  if (t == 0) qidx[b * MDIM] = 0;
  __syncthreads();
  for (int it = 1; it < MDIM; ++it) {
    float best = -1.f; int bi = 0;
    for (int i = t; i < NDIM; i += 256) {
      const float d = dist[i];
      if (d > best) { best = d; bi = i; }
    }
    rv[t] = best; ri[t] = bi;
    __syncthreads();
    for (int off = 128; off > 0; off >>= 1) {           // argmax, first-index ties
      if (t < off) {
        if (rv[t + off] > rv[t] || (rv[t + off] == rv[t] && ri[t + off] < ri[t])) {
          rv[t] = rv[t + off]; ri[t] = ri[t + off];
        }
      }
      __syncthreads();
    }
    const int sel = ri[0];
    if (t == 0) qidx[b * MDIM + it] = sel;
    const float sx = px[sel], sy = px[NDIM + sel], sz = px[2 * NDIM + sel];
    for (int i = t; i < NDIM; i += 256) {
      const float dx = px[i] - sx, dy = px[NDIM + i] - sy, dz = px[2 * NDIM + i] - sz;
      const float nd = dx * dx + dy * dy + dz * dz;
      if (nd < dist[i]) dist[i] = nd;
    }
    __syncthreads();
  }
  for (int m = t; m < MDIM; m += 256) {
    const int n = qidx[b * MDIM + m];
    qbuf[((size_t)b * 3 + 0) * MDIM + m] = px[n];
    qbuf[((size_t)b * 3 + 1) * MDIM + m] = px[NDIM + n];
    qbuf[((size_t)b * 3 + 2) * MDIM + m] = px[2 * NDIM + n];
  }
}

// ---------------------------------------------------------------- 5: KNN (top-32)
__global__ void knn_kernel(const float* __restrict__ x, const float* __restrict__ qbuf,
                           int* __restrict__ nidx) {
  const int gid = blockIdx.x * 256 + threadIdx.x;     // BMN threads
  const int b = gid >> 10, m = gid & (MDIM - 1);
  const float* px = x + (size_t)b * CDIM * NDIM;
  const float qx = qbuf[((size_t)b * 3 + 0) * MDIM + m];
  const float qy = qbuf[((size_t)b * 3 + 1) * MDIM + m];
  const float qz = qbuf[((size_t)b * 3 + 2) * MDIM + m];
  float bd[KNB]; int bi[KNB];
#pragma unroll
  for (int k = 0; k < KNB; ++k) { bd[k] = 3.4e38f; bi[k] = 0; }
  float worst = 3.4e38f; int wslot = 0;
  __shared__ float tx[256], ty[256], tz[256];
  for (int j0 = 0; j0 < NDIM; j0 += 256) {
    tx[threadIdx.x] = px[j0 + threadIdx.x];
    ty[threadIdx.x] = px[NDIM + j0 + threadIdx.x];
    tz[threadIdx.x] = px[2 * NDIM + j0 + threadIdx.x];
    __syncthreads();
    for (int t = 0; t < 256; ++t) {
      const float dx = qx - tx[t], dy = qy - ty[t], dz = qz - tz[t];
      const float d = dx * dx + dy * dy + dz * dz;
      if (d < worst) {
        bd[wslot] = d; bi[wslot] = j0 + t;
        worst = -1.f;
        for (int k = 0; k < KNB; ++k)
          if (bd[k] > worst) { worst = bd[k]; wslot = k; }
      }
    }
    __syncthreads();
  }
  for (int a = 1; a < KNB; ++a) {                      // ascending (d, idx)
    const float dv = bd[a]; const int iv = bi[a];
    int p = a - 1;
    while (p >= 0 && (bd[p] > dv || (bd[p] == dv && bi[p] > iv))) {
      bd[p + 1] = bd[p]; bi[p + 1] = bi[p]; --p;
    }
    bd[p + 1] = dv; bi[p + 1] = iv;
  }
  for (int k = 0; k < KNB; ++k) nidx[((size_t)b * KNB + k) * MDIM + m] = bi[k];
}

// ---------------------------------------------------------------- 6: gather + h0
__global__ void gather_kernel(const float* __restrict__ x, const float* __restrict__ qbuf,
                              const int* __restrict__ nidx, const float* __restrict__ sfin,
                              const float* ww1, const float* wb1,
                              _Float16* __restrict__ h0, float* __restrict__ relbuf,
                              float* __restrict__ wp1stat) {
  const size_t pt = (size_t)blockIdx.x * 256 + threadIdx.x;     // PTN
  const int k = pt & (KNB - 1);
  const size_t bm = pt >> 5;
  const int m = bm & (MDIM - 1), b = bm >> 10;
  const int n = nidx[((size_t)b * KNB + k) * MDIM + m];
  const float* px = x + (size_t)b * CDIM * NDIM;
  const float rx = px[n]            - qbuf[((size_t)b * 3 + 0) * MDIM + m];
  const float ry = px[NDIM + n]     - qbuf[((size_t)b * 3 + 1) * MDIM + m];
  const float rz = px[2 * NDIM + n] - qbuf[((size_t)b * 3 + 2) * MDIM + m];
  const float sg = sfin[b * NDIM + n];
  relbuf[pt] = rx; relbuf[PTN + pt] = ry; relbuf[2 * PTN + pt] = rz;
  _Float16* hp = h0 + pt * 64;                          // [pt][64] -> 128B/thread
  hp[0] = (_Float16)(rx * sg); hp[1] = (_Float16)(ry * sg); hp[2] = (_Float16)(rz * sg);
#pragma unroll
  for (int c = 3; c < 64; ++c) hp[c] = (_Float16)(px[(size_t)c * NDIM + n] * sg);
  __shared__ float ls[16];
  if (threadIdx.x < 16) ls[threadIdx.x] = 0.f;
  __syncthreads();
#pragma unroll
  for (int o = 0; o < 8; ++o) {
    const float p = wb1[o] + ww1[o * 3] * rx + ww1[o * 3 + 1] * ry + ww1[o * 3 + 2] * rz;
    atomicAdd(&ls[o], p);
    atomicAdd(&ls[8 + o], p * p);
  }
  __syncthreads();
  if (threadIdx.x < 16) atomicAdd(&wp1stat[threadIdx.x], ls[threadIdx.x]);
}

// ---------------------------------------------------------------- 7: f16 weight packs
__global__ void pack_kernel(const float* __restrict__ mw1, const float* __restrict__ mw2,
                            const float* __restrict__ lw,
                            _Float16* __restrict__ mw1h, _Float16* __restrict__ mw2h,
                            _Float16* __restrict__ wmatT) {
  const int id = blockIdx.x * 256 + threadIdx.x;        // 524288
  if (id < 64 * 64)  mw1h[id] = (_Float16)mw1[id];
  if (id < 128 * 64) mw2h[id] = (_Float16)mw2[id];
  // wmatT[o][w*128+c] = l_w[o, c, w]  -> contiguous-k B operand for final GEMM
  const int o = id >> 11, wc = id & 2047;
  const int c = wc & 127, w = wc >> 7;
  wmatT[id] = (_Float16)lw[((size_t)o * 128 + c) * 16 + w];
}

// ---------------------------------------------------------------- 8/10: m-net GEMM (WMMA)
// Out[pt][OUTC] = W[OUTC][64] * In[pt][64] + bias, per-channel stats via LDS+atomics.
template <int OUTC>
__global__ void mlp_gemm_wmma(const _Float16* __restrict__ Wh, const _Float16* __restrict__ Bin,
                              const float* __restrict__ bias, _Float16* __restrict__ Out,
                              float* __restrict__ stat) {
  const int lane = threadIdx.x & 31, wave = threadIdx.x >> 5;
  constexpr int ROWT = OUTC / 16;           // 4 or 8 row tiles of out-channels
  constexpr int COLW = 8 / ROWT;            // waves along points
  const int rowTile = wave % ROWT, colTile = wave / ROWT;
  const size_t ptBase = (size_t)blockIdx.x * (COLW * 16) + colTile * 16;
  const int kh = lane >> 4, nr = lane & 15;

  __shared__ _Float16 lw[OUTC * 64];        // weights staged once per block
  __shared__ float ls[2 * OUTC];
  for (int i = threadIdx.x; i < OUTC * 64; i += 256) lw[i] = Wh[i];
  for (int i = threadIdx.x; i < 2 * OUTC; i += 256) ls[i] = 0.f;
  __syncthreads();

  v8f acc = {};
#pragma unroll
  for (int kk = 0; kk < 2; ++kk) {          // 64 in-ch = 2 x K32
    const int kb = kk * 32;
    const _Float16* ar = &lw[(size_t)(rowTile * 16 + nr) * 64 + kb + kh * 16];
    const v16h a = load16h(ar, ar + 8);     // A: 16x32 out-ch x in-ch
    const _Float16* br = &Bin[(ptBase + nr) * 64 + kb + kh * 16];
    const v16h bm = load16h(br, br + 8);    // B: 32x16 in-ch x points
    acc = __builtin_amdgcn_wmma_f32_16x16x32_f16(false, a, false, bm,
                                                 (short)0, acc, false, false);
  }
  // C layout: elem r -> row(M)=kh*8+r (out-ch), col(N)=nr (point)
  half8 ov;
#pragma unroll
  for (int r = 0; r < 8; ++r) {
    const int row = rowTile * 16 + kh * 8 + r;
    const float v = acc[r] + bias[row];
    ov[r] = (_Float16)v;
    atomicAdd(&ls[row], v);
    atomicAdd(&ls[OUTC + row], v * v);
  }
  *(half8*)&Out[(ptBase + nr) * OUTC + rowTile * 16 + kh * 8] = ov;
  __syncthreads();
  for (int i = threadIdx.x; i < 2 * OUTC; i += 256) atomicAdd(&stat[i], ls[i]);
}

// ---------------------------------------------------------------- 9/11: BN+ReLU (+transpose)
__global__ void bnrelu_kernel(const _Float16* __restrict__ in, _Float16* __restrict__ out,
                              const float* __restrict__ stat, const float* __restrict__ g,
                              const float* __restrict__ bb, int Cc, float invCount,
                              int transposeOut, size_t total) {
  const size_t id = (size_t)blockIdx.x * 256 + threadIdx.x;
  if (id >= total) return;
  const int c = (int)(id % Cc);
  const size_t pt = id / Cc;
  const float mu = stat[c] * invCount;
  const float var = stat[Cc + c] * invCount - mu * mu;
  float v = (float)in[id];
  v = (v - mu) * rsqrtf(var + EPSF) * g[c] + bb[c];
  v = fmaxf(v, 0.f);
  out[transposeOut ? ((size_t)c * PTN + pt) : id] = (_Float16)v;
}

// ---------------------------------------------------------------- 12/13: weight net
__global__ void wnet2_kernel(const float* __restrict__ relbuf, const float* __restrict__ wp1stat,
                             const float* ww1, const float* wb1, const float* wg1, const float* wbb1,
                             const float* ww2, const float* wb2,
                             _Float16* __restrict__ wp2, float* __restrict__ wp2stat) {
  const size_t pt = (size_t)blockIdx.x * 256 + threadIdx.x;
  const float rx = relbuf[pt], ry = relbuf[PTN + pt], rz = relbuf[2 * PTN + pt];
  float h[8];
#pragma unroll
  for (int o = 0; o < 8; ++o) {
    const float pre = wb1[o] + ww1[o * 3] * rx + ww1[o * 3 + 1] * ry + ww1[o * 3 + 2] * rz;
    const float mu = wp1stat[o] * (1.f / (float)PTN);
    const float var = wp1stat[8 + o] * (1.f / (float)PTN) - mu * mu;
    h[o] = fmaxf((pre - mu) * rsqrtf(var + EPSF) * wg1[o] + wbb1[o], 0.f);
  }
  __shared__ float ls[16];
  if (threadIdx.x < 16) ls[threadIdx.x] = 0.f;
  __syncthreads();
#pragma unroll
  for (int o = 0; o < 8; ++o) {
    float p = wb2[o];
#pragma unroll
    for (int c = 0; c < 8; ++c) p += ww2[o * 8 + c] * h[c];
    wp2[(size_t)o * PTN + pt] = (_Float16)p;
    atomicAdd(&ls[o], p);
    atomicAdd(&ls[8 + o], p * p);
  }
  __syncthreads();
  if (threadIdx.x < 16) atomicAdd(&wp2stat[threadIdx.x], ls[threadIdx.x]);
}

__global__ void wnet3_kernel(const _Float16* __restrict__ wp2, const float* __restrict__ wp2stat,
                             const float* wg2, const float* wbb2,
                             const float* ww3, const float* wb3,
                             _Float16* __restrict__ wfin) {
  const size_t pt = (size_t)blockIdx.x * 256 + threadIdx.x;
  float h[8];
#pragma unroll
  for (int c = 0; c < 8; ++c) {
    const float mu = wp2stat[c] * (1.f / (float)PTN);
    const float var = wp2stat[8 + c] * (1.f / (float)PTN) - mu * mu;
    float v = (float)wp2[(size_t)c * PTN + pt];
    h[c] = fmaxf((v - mu) * rsqrtf(var + EPSF) * wg2[c] + wbb2[c], 0.f);
  }
#pragma unroll
  for (int o = 0; o < 16; ++o) {
    float p = wb3[o];
#pragma unroll
    for (int c = 0; c < 8; ++c) p += ww3[o * 8 + c] * h[c];
    wfin[(size_t)o * PTN + pt] = (_Float16)p;   // [16][pt]: A operand for agg
  }
}

// ---------------------------------------------------------------- 14: agg (one WMMA per tile)
// agg[bm][w*128+c] = sum_k wfin[w][bm,k] * mh2[c][bm,k]; K=32 == the WMMA k-dim.
__global__ void agg_wmma_kernel(const _Float16* __restrict__ wfin,
                                const _Float16* __restrict__ mh2,
                                _Float16* __restrict__ aggbuf) {
  const int lane = threadIdx.x & 31, wave = threadIdx.x >> 5;
  const size_t bm = (size_t)blockIdx.x * 8 + wave;
  const size_t base = bm * KNB;
  const int kh = lane >> 4, nr = lane & 15;
  const _Float16* ar = &wfin[(size_t)nr * PTN + base + kh * 16];
  const v16h a = load16h(ar, ar + 8);                   // 16(w) x 32(k)
#pragma unroll
  for (int ct = 0; ct < 8; ++ct) {                      // 128 c-channels
    const _Float16* br = &mh2[(size_t)(ct * 16 + nr) * PTN + base + kh * 16];
    const v16h bmv = load16h(br, br + 8);               // 32(k) x 16(c)
    v8f acc = {};
    acc = __builtin_amdgcn_wmma_f32_16x16x32_f16(false, a, false, bmv,
                                                 (short)0, acc, false, false);
#pragma unroll
    for (int r = 0; r < 8; ++r) {
      const int wch = kh * 8 + r;
      aggbuf[bm * 2048 + (size_t)wch * 128 + ct * 16 + nr] = (_Float16)acc[r];
    }
  }
}

// ---------------------------------------------------------------- 15: final GEMM (WMMA)
// Y[4096x256] = agg[4096x2048] * wmatT^T ; block = 32 rows x 64 cols, 8 waves.
__global__ void final_gemm_wmma(const _Float16* __restrict__ aggbuf,
                                const _Float16* __restrict__ wmatT,
                                const float* __restrict__ lb,
                                float* __restrict__ ybuf, float* __restrict__ ystat) {
  const int lane = threadIdx.x & 31, wave = threadIdx.x >> 5;
  const int rowTile = wave & 1, colTile = wave >> 1;
  const size_t row0 = (size_t)blockIdx.x * 32 + rowTile * 16;
  const int col0 = blockIdx.y * 64 + colTile * 16;
  const int kh = lane >> 4, nr = lane & 15;
  __shared__ float ls[128];
  for (int i = threadIdx.x; i < 128; i += 256) ls[i] = 0.f;
  __syncthreads();

  const _Float16* arow = &aggbuf[(row0 + nr) * 2048 + kh * 16];
  const _Float16* brow = &wmatT[(size_t)(col0 + nr) * 2048 + kh * 16];
  v8f acc = {};
#pragma unroll 4
  for (int kb = 0; kb < 2048; kb += 32) {
    __builtin_prefetch(arow + kb + 256, 0, 0);          // global_prefetch next tiles
    __builtin_prefetch(brow + kb + 256, 0, 0);
    const v16h a = load16h(arow + kb, arow + kb + 8);   // 16(row) x 32(k)
    const v16h b = load16h(brow + kb, brow + kb + 8);   // 32(k) x 16(col)
    acc = __builtin_amdgcn_wmma_f32_16x16x32_f16(false, a, false, b,
                                                 (short)0, acc, false, false);
  }
  const int col = col0 + nr;
#pragma unroll
  for (int r = 0; r < 8; ++r) {
    const size_t row = row0 + kh * 8 + r;
    const float v = acc[r] + lb[col];
    ybuf[row * 256 + col] = v;
    atomicAdd(&ls[colTile * 16 + nr], v);
    atomicAdd(&ls[64 + colTile * 16 + nr], v * v);
  }
  __syncthreads();
  for (int i = threadIdx.x; i < 128; i += 256) {
    const int lc = i & 63, part = i >> 6;
    atomicAdd(&ystat[part * 256 + blockIdx.y * 64 + lc], ls[i]);
  }
}

// ---------------------------------------------------------------- 16: output
__global__ void output_kernel(const float* __restrict__ qbuf, const float* __restrict__ ybuf,
                              const float* __restrict__ ystat,
                              const float* lg, const float* lbb, float* __restrict__ out) {
  const size_t id = (size_t)blockIdx.x * 256 + threadIdx.x;   // B*259*M
  const int m = (int)(id & (MDIM - 1));
  const size_t t = id >> 10;
  const int ch = (int)(t % 259), b = (int)(t / 259);
  if (ch < 3) {
    out[id] = qbuf[((size_t)b * 3 + ch) * MDIM + m];
  } else {
    const int o = ch - 3;
    const float mu = ystat[o] * (1.f / BMN);
    const float var = ystat[256 + o] * (1.f / BMN) - mu * mu;
    float v = ybuf[((size_t)b * MDIM + m) * 256 + o];
    v = (v - mu) * rsqrtf(var + EPSF) * lg[o] + lbb[o];
    out[id] = fmaxf(v, 0.f);
  }
}

// ---------------------------------------------------------------- host launch
extern "C" void kernel_launch(void* const* d_in, const int* in_sizes, int n_in,
                              void* d_out, int out_size, void* d_ws, size_t ws_size,
                              hipStream_t stream) {
  const float* x    = (const float*)d_in[0];
  const float* sw1  = (const float*)d_in[1],  *sb1 = (const float*)d_in[2];
  const float* sg1  = (const float*)d_in[3],  *sbb1 = (const float*)d_in[4];
  const float* sw2  = (const float*)d_in[5],  *sb2 = (const float*)d_in[6];
  const float* sg2  = (const float*)d_in[7],  *sbb2 = (const float*)d_in[8];
  const float* sw3  = (const float*)d_in[9],  *sb3 = (const float*)d_in[10];
  const float* ww1  = (const float*)d_in[11], *wb1 = (const float*)d_in[12];
  const float* wg1  = (const float*)d_in[13], *wbb1 = (const float*)d_in[14];
  const float* ww2  = (const float*)d_in[15], *wb2 = (const float*)d_in[16];
  const float* wg2  = (const float*)d_in[17], *wbb2 = (const float*)d_in[18];
  const float* ww3  = (const float*)d_in[19], *wb3 = (const float*)d_in[20];
  const float* mw1  = (const float*)d_in[21], *mb1 = (const float*)d_in[22];
  const float* mg1  = (const float*)d_in[23], *mbb1 = (const float*)d_in[24];
  const float* mw2  = (const float*)d_in[25], *mb2 = (const float*)d_in[26];
  const float* mg2  = (const float*)d_in[27], *mbb2 = (const float*)d_in[28];
  const float* lw   = (const float*)d_in[29], *lb  = (const float*)d_in[30];
  const float* lg   = (const float*)d_in[31], *lbb = (const float*)d_in[32];

  char* ws = (char*)d_ws;
  // all offsets 4 KiB aligned
  constexpr size_t OFF_STAT = 0;
  constexpr size_t OFF_SRAW = 4096;
  constexpr size_t OFF_P2   = OFF_SRAW + (size_t)NP * 4;            // 512 KB
  constexpr size_t OFF_SFIN = OFF_P2   + (size_t)NP * 8 * 4;
  constexpr size_t OFF_QIDX = OFF_SFIN + (size_t)NP * 4;
  constexpr size_t OFF_Q    = OFF_QIDX + (size_t)BMN * 4;
  constexpr size_t OFF_NIDX = OFF_Q    + (size_t)BDIM * 3 * MDIM * 4;
  constexpr size_t OFF_REL  = OFF_NIDX + (size_t)BDIM * KNB * MDIM * 4;
  constexpr size_t OFF_H0   = OFF_REL  + 3 * PTN * 4;
  constexpr size_t OFF_MW1H = OFF_H0   + 64 * PTN * 2;
  constexpr size_t OFF_MW2H = OFF_MW1H + 8192;
  constexpr size_t OFF_WMAT = OFF_MW2H + 16384;
  constexpr size_t OFF_MP1  = OFF_WMAT + 2048ull * 256 * 2;
  constexpr size_t OFF_MH1  = OFF_MP1  + 64 * PTN * 2;
  constexpr size_t OFF_MP2  = OFF_MH1  + 64 * PTN * 2;
  constexpr size_t OFF_MH2  = OFF_MP2  + 128 * PTN * 2;
  constexpr size_t OFF_WP2  = OFF_MH2  + 128 * PTN * 2;
  constexpr size_t OFF_WFIN = OFF_WP2  + 8 * PTN * 2;
  constexpr size_t OFF_AGG  = OFF_WFIN + 16 * PTN * 2;
  constexpr size_t OFF_Y    = OFF_AGG  + (size_t)BMN * 2048 * 2;

  float*     st    = (float*)(ws + OFF_STAT);
  float*     sraw  = (float*)(ws + OFF_SRAW);
  float*     p2b   = (float*)(ws + OFF_P2);
  float*     sfin  = (float*)(ws + OFF_SFIN);
  int*       qidx  = (int*)  (ws + OFF_QIDX);
  float*     qbuf  = (float*)(ws + OFF_Q);
  int*       nidx  = (int*)  (ws + OFF_NIDX);
  float*     relb  = (float*)(ws + OFF_REL);
  _Float16*  h0    = (_Float16*)(ws + OFF_H0);
  _Float16*  mw1h  = (_Float16*)(ws + OFF_MW1H);
  _Float16*  mw2h  = (_Float16*)(ws + OFF_MW2H);
  _Float16*  wmatT = (_Float16*)(ws + OFF_WMAT);
  _Float16*  mp1   = (_Float16*)(ws + OFF_MP1);
  _Float16*  mh1   = (_Float16*)(ws + OFF_MH1);
  _Float16*  mp2   = (_Float16*)(ws + OFF_MP2);
  _Float16*  mh2   = (_Float16*)(ws + OFF_MH2);
  _Float16*  wp2   = (_Float16*)(ws + OFF_WP2);
  _Float16*  wfin  = (_Float16*)(ws + OFF_WFIN);
  _Float16*  aggb  = (_Float16*)(ws + OFF_AGG);
  float*     ybuf  = (float*)(ws + OFF_Y);

  zero_stats_kernel<<<1, 256, 0, stream>>>(st);
  kde_kernel<<<dim3(NDIM / 256, BDIM), 256, 0, stream>>>(x, sraw, st);
  smlp1_kernel<<<NP / 256, 256, 0, stream>>>(sraw, st, sw1, sb1, sg1, sbb1, sw2, sb2,
                                             p2b, st + ST_P2);
  smlp2_kernel<<<NP / 256, 256, 0, stream>>>(p2b, st + ST_P2, sg2, sbb2, sw3, sb3, sfin);
  fps_kernel<<<BDIM, 256, 0, stream>>>(x, qidx, qbuf);
  knn_kernel<<<BMN / 256, 256, 0, stream>>>(x, qbuf, nidx);
  gather_kernel<<<(int)(PTN / 256), 256, 0, stream>>>(x, qbuf, nidx, sfin, ww1, wb1,
                                                      h0, relb, st + ST_WP1);
  pack_kernel<<<2048, 256, 0, stream>>>(mw1, mw2, lw, mw1h, mw2h, wmatT);
  mlp_gemm_wmma<64><<<(int)(PTN / 32), 256, 0, stream>>>(mw1h, h0, mb1, mp1, st + ST_MP1);
  bnrelu_kernel<<<(int)(64 * PTN / 256), 256, 0, stream>>>(mp1, mh1, st + ST_MP1, mg1, mbb1,
                                                           64, 1.f / (float)PTN, 0, 64 * PTN);
  mlp_gemm_wmma<128><<<(int)(PTN / 16), 256, 0, stream>>>(mw2h, mh1, mb2, mp2, st + ST_MP2);
  bnrelu_kernel<<<(int)(128 * PTN / 256), 256, 0, stream>>>(mp2, mh2, st + ST_MP2, mg2, mbb2,
                                                            128, 1.f / (float)PTN, 1, 128 * PTN);
  wnet2_kernel<<<(int)(PTN / 256), 256, 0, stream>>>(relb, st + ST_WP1, ww1, wb1, wg1, wbb1,
                                                     ww2, wb2, wp2, st + ST_WP2);
  wnet3_kernel<<<(int)(PTN / 256), 256, 0, stream>>>(wp2, st + ST_WP2, wg2, wbb2, ww3, wb3, wfin);
  agg_wmma_kernel<<<BMN / 8, 256, 0, stream>>>(wfin, mh2, aggb);
  final_gemm_wmma<<<dim3(BMN / 32, 4), 256, 0, stream>>>(aggb, wmatT, lb, ybuf, st + ST_Y);
  output_kernel<<<(BDIM * 259 * MDIM) / 256, 256, 0, stream>>>(qbuf, ybuf, st + ST_Y,
                                                               lg, lbb, (float*)d_out);
  (void)in_sizes; (void)n_in; (void)out_size; (void)ws_size;
}
